// SelfAttention2d_59820304499407
// MI455X (gfx1250) — matrix operand order repaired and checked by reading, not compiled, not run on hardware
//
#include <hip/hip_runtime.h>
#include <hip/hip_bf16.h>

// ---------------- problem constants ----------------
constexpr int BATCH = 2;
constexpr int C_IN  = 256;
constexpr int HH    = 48;
constexpr int WW    = 48;
constexpr int S     = HH * WW;      // 2304
constexpr int HEADS = 8;
constexpr int DH    = 64;
constexpr int INNER = HEADS * DH;   // 512
constexpr int OIN   = 3 * INNER;    // 1536
constexpr float LOG_SCALE_MAX = 4.6051702f;  // ln(100)

typedef __attribute__((ext_vector_type(16))) _Float16 v16h;
typedef __attribute__((ext_vector_type(8)))  _Float16 v8h;
typedef __attribute__((ext_vector_type(8)))  float    v8f;
typedef __attribute__((ext_vector_type(4)))  float    v4f;
typedef __attribute__((ext_vector_type(4)))  unsigned v4u;
typedef __attribute__((ext_vector_type(8)))  int      v8i;
typedef __attribute__((ext_vector_type(4)))  int      v4i;

#if defined(__has_builtin)
#if __has_builtin(__builtin_amdgcn_tensor_load_to_lds) && __has_builtin(__builtin_amdgcn_s_wait_tensorcnt)
#define USE_TDM 1
#endif
#endif
#ifndef USE_TDM
#define USE_TDM 0
#endif

__device__ __forceinline__ v16h mkv16(v8h lo, v8h hi) {
  v16h r;
#pragma unroll
  for (int i = 0; i < 8; ++i) { r[i] = lo[i]; r[i + 8] = hi[i]; }
  return r;
}
__device__ __forceinline__ v8h ldv8(const _Float16* p) { return *(const v8h*)p; }
__device__ __forceinline__ v8f zero8f() {
  v8f z;
#pragma unroll
  for (int i = 0; i < 8; ++i) z[i] = 0.0f;
  return z;
}

#if USE_TDM
__device__ __forceinline__ unsigned lds_addr_of(void* p) {
  return (unsigned)(uintptr_t)(__attribute__((address_space(3))) void*)p;
}
// 2-D f16 tile load: nrows rows of 64 halves, row stride 64 halves in memory,
// LDS rows padded to 72 halves via TDM pad (interval=32 DW, amount=4 DW).
__device__ __forceinline__ void tdm_load_tile_f16(unsigned lds_byte_addr,
                                                  const _Float16* g, int nrows) {
  unsigned long long ga = (unsigned long long)(uintptr_t)g;
  v4u g0;
  g0[0] = 1u;                                   // count=1, user mode
  g0[1] = lds_byte_addr;                        // lds_addr
  g0[2] = (unsigned)ga;                         // global_addr[31:0]
  g0[3] = (unsigned)((ga >> 32) & 0x1ffffffull) // global_addr[56:32]
        | (2u << 30);                           // type=2 (image)
  v8i g1;
  g1[0] = (int)((1u << 16)      // data_size = 2 bytes
              | (1u << 20)      // pad_enable
              | (4u << 22)      // pad_interval: 32 DWORDs (one 64-half row)
              | (3u << 25));    // pad_amount: 4 DWORDs (8 halves) -> stride 72
  g1[1] = (int)(64u << 16);                     // tensor_dim0 = 64 (lo16)
  g1[2] = (int)((unsigned)S << 16);             // dim0 hi16=0 | tensor_dim1 lo16 = S
  g1[3] = (int)(64u << 16);                     // dim1 hi16=0 | tile_dim0 = 64
  g1[4] = (int)(unsigned)nrows;                 // tile_dim1 | tile_dim2 = 0
  g1[5] = 64;                                   // tensor_dim0_stride lo32
  g1[6] = 0;
  g1[7] = 0;
  v4i zz; zz[0] = 0; zz[1] = 0; zz[2] = 0; zz[3] = 0;
  v8i z8;
#pragma unroll
  for (int i = 0; i < 8; ++i) z8[i] = 0;
  __builtin_amdgcn_tensor_load_to_lds(g0, g1, zz, zz, z8, 0);
}
#endif

// ============================================================
// Shared 64x64 GEMM core: D(64x64) += A(64xK,f32) * B
// 128 threads = 4 waves, wave w owns rows w*16..w*16+15.
// BNK=false: B is [K][N] (transposed during staging)
// BNK=true : B is [N][K] f16 row-major (pure vector staging)
// ============================================================
template <typename TB, bool BNK = false>
__device__ __forceinline__ void gemm_core_64x64(
    const float* __restrict__ A, const TB* __restrict__ Bm,
    int lda, int ldb, int K, int m0, int n0,
    _Float16* lds_a, _Float16* lds_bt, v8f acc[4]) {
  const int tid  = threadIdx.x;
  const int lane = tid & 31;
  const int wave = tid >> 5;
  const int hi   = lane >> 4;
  const int l16  = lane & 15;
#pragma unroll
  for (int t = 0; t < 4; ++t) acc[t] = zero8f();

  for (int k0 = 0; k0 < K; k0 += 32) {
    __syncthreads();
    if (k0 + 32 < K) {  // speculative prefetch of next B tile
      if constexpr (BNK)
        __builtin_prefetch((const _Float16*)Bm + (size_t)n0 * ldb + k0 + 32, 0, 0);
      else
        __builtin_prefetch(Bm + (size_t)(k0 + 32) * ldb + n0, 0, 0);
    }

    // A tile 64x32: 256 chunks of 8 floats, 2 per thread, vector ld/st
#pragma unroll
    for (int j = 0; j < 2; ++j) {
      int cc = tid + j * 128;
      int m = cc >> 2, ks = (cc & 3) * 8;
      const float* ap = A + (size_t)(m0 + m) * lda + k0 + ks;
      v4f f0 = *(const v4f*)ap;
      v4f f1 = *(const v4f*)(ap + 4);
      v8h hv;
#pragma unroll
      for (int e = 0; e < 4; ++e) { hv[e] = (_Float16)f0[e]; hv[e + 4] = (_Float16)f1[e]; }
      *(v8h*)&lds_a[m * 40 + ks] = hv;
    }
    if constexpr (BNK) {
      // B[n][k] f16: straight vector copy into lds_bt[n][k]
      int n = tid >> 1, koff = (tid & 1) * 16;
      const _Float16* bp = (const _Float16*)Bm + (size_t)(n0 + n) * ldb + k0 + koff;
      *(v8h*)&lds_bt[n * 40 + koff]     = ldv8(bp);
      *(v8h*)&lds_bt[n * 40 + koff + 8] = ldv8(bp + 8);
    } else {
      // B[k][n] -> lds_bt[n][k]: 256 chunks of 8 along n, 2 per thread
#pragma unroll
      for (int j = 0; j < 2; ++j) {
        int cc = tid + j * 128;
        int k = cc >> 3, n8 = (cc & 7) * 8;
        if constexpr (sizeof(TB) == 2) {
          v8h hv = *(const v8h*)((const _Float16*)Bm + (size_t)(k0 + k) * ldb + n0 + n8);
#pragma unroll
          for (int e = 0; e < 8; ++e) lds_bt[(n8 + e) * 40 + k] = hv[e];
        } else {
          const float* bp = (const float*)Bm + (size_t)(k0 + k) * ldb + n0 + n8;
          v4f f0 = *(const v4f*)bp;
          v4f f1 = *(const v4f*)(bp + 4);
#pragma unroll
          for (int e = 0; e < 4; ++e) {
            lds_bt[(n8 + e) * 40 + k]     = (_Float16)f0[e];
            lds_bt[(n8 + 4 + e) * 40 + k] = (_Float16)f1[e];
          }
        }
      }
    }
    __syncthreads();

    const _Float16* ar = lds_a + (wave * 16 + l16) * 40;
    v16h af = mkv16(ldv8(ar + hi * 16), ldv8(ar + hi * 16 + 8));
#pragma unroll
    for (int t = 0; t < 4; ++t) {
      const _Float16* br = lds_bt + (t * 16 + l16) * 40 + hi * 16;
      v16h bf = mkv16(ldv8(br), ldv8(br + 8));
      acc[t] = __builtin_amdgcn_wmma_f32_16x16x32_f16(
          false, af, false, bf, (short)0, acc[t], false, false);
    }
  }
}

// ============================================================
// Kernel 1: proj_in  y[o,s] = sum_c w_in[o,c] x[b,c,s] + b_in[o]
// q,k -> [bh][s][64] f16 (raw; k normalized later, q in k_attn)
// v   -> [bh][64][s] f16 (pre-transposed for the attention PV stage)
// A 64-row M-tile lies inside one q/k/v segment of one head, so the
// target buffer is block-uniform and q/k stores are v8h vectors.
// ============================================================
__global__ __launch_bounds__(128) void k_proj_in(
    const float* __restrict__ x, const float* __restrict__ w_in,
    const float* __restrict__ b_in,
    _Float16* __restrict__ qn, _Float16* __restrict__ kn,
    _Float16* __restrict__ vv) {
  const int MT = OIN / 64, NT = S / 64;               // 24, 36
  int blk = blockIdx.x;
  int b   = blk / (MT * NT);
  int r   = blk % (MT * NT);
  int m0  = (r / NT) * 64;
  int n0  = (r % NT) * 64;
  const float* Bm = x + (size_t)b * C_IN * S;

  __shared__ alignas(16) _Float16 lds_a[64 * 40];
  __shared__ alignas(16) _Float16 lds_bt[64 * 40];
  v8f acc[4];
  gemm_core_64x64<float>(w_in, Bm, C_IN, S, C_IN, m0, n0, lds_a, lds_bt, acc);

  const int lane = threadIdx.x & 31;
  const int wave = threadIdx.x >> 5;
  const int hi = lane >> 4, l16 = lane & 15;
  const int head = m0 / 192;            // uniform over the tile
  const int seg  = (m0 % 192) / 64;     // 0=q 1=k 2=v, uniform
  const int d0   = wave * 16 + hi * 8;
  const int o0   = m0 + d0;
  const int bh   = b * HEADS + head;
#pragma unroll
  for (int t = 0; t < 4; ++t) {
    int s = n0 + t * 16 + l16;
    if (seg < 2) {
      _Float16* dst = (seg == 0 ? qn : kn) + ((size_t)bh * S + s) * DH + d0;
      v8h hv;
#pragma unroll
      for (int rr = 0; rr < 8; ++rr) hv[rr] = (_Float16)(acc[t][rr] + b_in[o0 + rr]);
      *(v8h*)dst = hv;
    } else {
      _Float16* dst = vv + ((size_t)bh * DH + d0) * S + s;   // [bh][d][s]
#pragma unroll
      for (int rr = 0; rr < 8; ++rr)
        dst[(size_t)rr * S] = (_Float16)(acc[t][rr] + b_in[o0 + rr]);
    }
  }
}

// ============================================================
// Kernel 2: in-place L2 normalization of K rows (q is normalized
// inside the attention kernel).  One wave per 64-elem row.
// ============================================================
__global__ __launch_bounds__(256) void k_l2norm_k(_Float16* __restrict__ kn) {
  int row  = blockIdx.x * 8 + (threadIdx.x >> 5);
  int lane = threadIdx.x & 31;
  _Float16* p = kn + (size_t)row * DH;
  float a = (float)p[lane], c = (float)p[lane + 32];
  float ss = a * a + c * c;
#pragma unroll
  for (int off = 1; off < 32; off <<= 1) ss += __shfl_xor(ss, off, 32);
  float inv = 1.0f / fmaxf(sqrtf(ss), 1e-12f);
  p[lane]      = (_Float16)(a * inv);
  p[lane + 32] = (_Float16)(c * inv);
}

// ============================================================
// Kernel 3: flash attention per (b,head).  Block = 4 waves, each
// wave owns a 16-row query tile; key tiles of 32.  K tiles arrive
// via TDM (double buffered); V copied from the pre-transposed
// [bh][d][s] buffer with pure vector moves.
// ============================================================
__global__ __launch_bounds__(128) void k_attn(
    const _Float16* __restrict__ qn, const _Float16* __restrict__ kn,
    const _Float16* __restrict__ vv, const float* __restrict__ log_scale,
    float* __restrict__ ao) {
  const int QT = S / 64;                        // 36
  int blk  = blockIdx.x;
  int bh   = blk / QT;
  int qt   = blk % QT;
  int head = bh % HEADS;
  float scale = __expf(fminf(log_scale[head], LOG_SCALE_MAX));

  const _Float16* qb = qn + (size_t)bh * S * DH;
  const _Float16* kb = kn + (size_t)bh * S * DH;
  const _Float16* vb = vv + (size_t)bh * DH * S;   // [d][s]

  const int tid = threadIdx.x, lane = tid & 31, wave = tid >> 5;
  const int hi = lane >> 4, l16 = lane & 15;
  const int q0 = qt * 64 + wave * 16;

  __shared__ alignas(16) _Float16 lds_k[2][32 * 72];  // [buf][key][d] (stride 72)
  __shared__ alignas(16) _Float16 lds_vt[64 * 40];    // [d][key]
  __shared__ alignas(16) _Float16 lds_p[4 * 16 * 32];

  // Q fragments (d 0..31 and 32..63) + fused L2 normalization
  const _Float16* qrow = qb + (size_t)(q0 + l16) * DH;
  v16h qf0 = mkv16(ldv8(qrow + hi * 16),      ldv8(qrow + hi * 16 + 8));
  v16h qf1 = mkv16(ldv8(qrow + 32 + hi * 16), ldv8(qrow + 40 + hi * 16));
  {
    float ss = 0.0f;
#pragma unroll
    for (int e = 0; e < 16; ++e) {
      float a = (float)qf0[e], c = (float)qf1[e];
      ss += a * a + c * c;
    }
    ss += __shfl_xor(ss, 16, 32);   // partner lane holds the other 32 of 64 elems
    float inv = 1.0f / fmaxf(sqrtf(ss), 1e-12f);
#pragma unroll
    for (int e = 0; e < 16; ++e) {
      qf0[e] = (_Float16)((float)qf0[e] * inv);
      qf1[e] = (_Float16)((float)qf1[e] * inv);
    }
  }

  float m[8], l[8];
  v8f oacc[4];
#pragma unroll
  for (int r = 0; r < 8; ++r) { m[r] = -1e30f; l[r] = 0.0f; }
#pragma unroll
  for (int t = 0; t < 4; ++t) oacc[t] = zero8f();

  const int NKT = S / 32;   // 72 key tiles
  int cur = 0;
#if USE_TDM
  if (wave == 0) tdm_load_tile_f16(lds_addr_of(&lds_k[0][0]), kb, 32);
#endif

  for (int it = 0; it < NKT; ++it) {
    const int k0 = it * 32;
    __syncthreads();                     // everyone done with previous tile
#if USE_TDM
    if (wave == 0) {
      if (it + 1 < NKT) {                // prefetch next K tile into other buffer
        tdm_load_tile_f16(lds_addr_of(&lds_k[cur ^ 1][0]),
                          kb + (size_t)(k0 + 32) * DH, 32);
        __builtin_amdgcn_s_wait_tensorcnt(1);   // current tile complete
      } else {
        __builtin_amdgcn_s_wait_tensorcnt(0);
      }
    }
#else
    {   // manual K staging
      int key = tid >> 2, seg = tid & 3;
      const _Float16* kr = kb + (size_t)(k0 + key) * DH + seg * 16;
      *(v8h*)&lds_k[0][key * 72 + seg * 16]     = ldv8(kr);
      *(v8h*)&lds_k[0][key * 72 + seg * 16 + 8] = ldv8(kr + 8);
    }
#endif
    {   // stage V tile [d][key]: straight vector copy from [bh][d][s]
      int d = tid >> 1, koff = (tid & 1) * 16;
      const _Float16* vr = vb + (size_t)d * S + k0 + koff;
      *(v8h*)&lds_vt[d * 40 + koff]     = ldv8(vr);
      *(v8h*)&lds_vt[d * 40 + koff + 8] = ldv8(vr + 8);
    }
    __syncthreads();
    const _Float16* kt = &lds_k[cur][0];

    // S = Q K^T  (two 16x16 key tiles, d reduced over 2 WMMA steps)
    v8f s0 = zero8f(), s1 = zero8f();
#pragma unroll
    for (int ks = 0; ks < 2; ++ks) {
      v16h qf = ks ? qf1 : qf0;
      const _Float16* b0 = kt + (0 * 16 + l16) * 72 + ks * 32 + hi * 16;
      const _Float16* b1 = kt + (1 * 16 + l16) * 72 + ks * 32 + hi * 16;
      v16h bf0 = mkv16(ldv8(b0), ldv8(b0 + 8));
      v16h bf1 = mkv16(ldv8(b1), ldv8(b1 + 8));
      s0 = __builtin_amdgcn_wmma_f32_16x16x32_f16(false, qf, false, bf0, (short)0, s0, false, false);
      s1 = __builtin_amdgcn_wmma_f32_16x16x32_f16(false, qf, false, bf1, (short)0, s1, false, false);
    }

    // online softmax (per-row stats live in C-fragment row layout)
    float rm[8];
#pragma unroll
    for (int r = 0; r < 8; ++r) {
      s0[r] *= scale; s1[r] *= scale;
      rm[r] = fmaxf(s0[r], s1[r]);
    }
#pragma unroll
    for (int off = 1; off < 16; off <<= 1)
#pragma unroll
      for (int r = 0; r < 8; ++r) rm[r] = fmaxf(rm[r], __shfl_xor(rm[r], off, 32));

    float p0[8], p1[8], rs[8];
#pragma unroll
    for (int r = 0; r < 8; ++r) {
      float mn = fmaxf(m[r], rm[r]);
      float al = __expf(m[r] - mn);
      m[r] = mn;
      p0[r] = __expf(s0[r] - mn);
      p1[r] = __expf(s1[r] - mn);
      rs[r] = p0[r] + p1[r];
      l[r] *= al;
#pragma unroll
      for (int t = 0; t < 4; ++t) oacc[t][r] *= al;
    }
#pragma unroll
    for (int off = 1; off < 16; off <<= 1)
#pragma unroll
      for (int r = 0; r < 8; ++r) rs[r] += __shfl_xor(rs[r], off, 32);
#pragma unroll
    for (int r = 0; r < 8; ++r) l[r] += rs[r];

    // re-layout P through per-wave LDS into an A fragment
    _Float16* pp = lds_p + wave * (16 * 32);
#pragma unroll
    for (int r = 0; r < 8; ++r) {
      pp[(hi * 8 + r) * 32 + l16]      = (_Float16)p0[r];
      pp[(hi * 8 + r) * 32 + 16 + l16] = (_Float16)p1[r];
    }
    __builtin_amdgcn_wave_barrier();   // keep store->load ordering (LDS in-order per wave)
    const _Float16* pr = pp + l16 * 32;
    v16h pf = mkv16(ldv8(pr + hi * 16), ldv8(pr + hi * 16 + 8));

    // O += P * V
#pragma unroll
    for (int t = 0; t < 4; ++t) {
      const _Float16* vr2 = &lds_vt[(t * 16 + l16) * 40 + hi * 16];
      v16h vf = mkv16(ldv8(vr2), ldv8(vr2 + 8));
      oacc[t] = __builtin_amdgcn_wmma_f32_16x16x32_f16(
          false, pf, false, vf, (short)0, oacc[t], false, false);
    }
#if USE_TDM
    cur ^= 1;
#endif
  }

  // epilogue: O /= l, write [bh][row][d] f32
#pragma unroll
  for (int t = 0; t < 4; ++t) {
#pragma unroll
    for (int r = 0; r < 8; ++r) {
      int row = q0 + hi * 8 + r;
      int d   = t * 16 + l16;
      ao[((size_t)bh * S + row) * DH + d] = oacc[t][r] / l[r];
    }
  }
}

// ============================================================
// Kernel 4: depthwise 3x3 (SAME), merging heads on the fly.
// Channel-fastest thread order: coalesced ao reads and dw writes.
// out layout: dw[b][s][ch] f16
// ============================================================
__global__ __launch_bounds__(256) void k_dw(
    const float* __restrict__ ao, const float* __restrict__ w_dw,
    _Float16* __restrict__ dwo) {
  int i = blockIdx.x * 256 + threadIdx.x;
  const int TOT = BATCH * INNER * S;
  if (i >= TOT) return;
  int ch = i % INNER;
  int t  = i / INNER;
  int s = t % S, b = t / S;
  int y = s / WW, x = s % WW;
  int head = ch >> 6, d = ch & 63;
  const float* base = ao + (size_t)(b * HEADS + head) * S * DH + d;
  float acc = 0.0f;
#pragma unroll
  for (int ky = 0; ky < 3; ++ky) {
    int yy = y + ky - 1;
    if (yy < 0 || yy >= HH) continue;
#pragma unroll
    for (int kx = 0; kx < 3; ++kx) {
      int xx = x + kx - 1;
      if (xx < 0 || xx >= WW) continue;
      acc += w_dw[ch * 9 + ky * 3 + kx] * base[(size_t)(yy * WW + xx) * DH];
    }
  }
  dwo[i] = (_Float16)acc;   // i == (b*S + s)*INNER + ch
}

// ============================================================
// Kernel 5: pointwise 1x1  pw[s,o] = sum_ch w_pw[o,ch] dw[s,ch]
// B supplied row-major [n=s][k=ch] (BNK path); output [b][s][o].
// ============================================================
__global__ __launch_bounds__(128) void k_pw(
    const _Float16* __restrict__ dwi, const float* __restrict__ w_pw,
    float* __restrict__ pw) {
  const int MT = INNER / 64, NT = S / 64;   // 8, 36
  int blk = blockIdx.x;
  int b   = blk / (MT * NT);
  int r   = blk % (MT * NT);
  int m0  = (r / NT) * 64;
  int n0  = (r % NT) * 64;
  const _Float16* Bm = dwi + (size_t)b * S * INNER;  // [s][ch]

  __shared__ alignas(16) _Float16 lds_a[64 * 40];
  __shared__ alignas(16) _Float16 lds_bt[64 * 40];
  v8f acc[4];
  gemm_core_64x64<_Float16, true>(w_pw, Bm, INNER, INNER, INNER, m0, n0,
                                  lds_a, lds_bt, acc);

  const int lane = threadIdx.x & 31;
  const int wave = threadIdx.x >> 5;
  const int hi = lane >> 4, l16 = lane & 15;
  float* pb = pw + (size_t)b * S * INNER;
#pragma unroll
  for (int t = 0; t < 4; ++t) {
    int s = n0 + t * 16 + l16;
    float* dst = pb + (size_t)s * INNER + m0 + wave * 16 + hi * 8;
    v4f lo, hi4;
#pragma unroll
    for (int e = 0; e < 4; ++e) { lo[e] = acc[t][e]; hi4[e] = acc[t][e + 4]; }
    *(v4f*)dst = lo;
    *(v4f*)(dst + 4) = hi4;
  }
}

// ============================================================
// Kernel 6: channel LayerNorm (512) + split + modulate residual
// pw is [b][s][512] so the reduction reads are fully coalesced.
// ============================================================
__global__ __launch_bounds__(256) void k_ln(
    const float* __restrict__ pw, const float* __restrict__ x,
    const float* __restrict__ gamma, const float* __restrict__ beta,
    float* __restrict__ out) {
  int pix = blockIdx.x % S;
  int b   = blockIdx.x / S;
  int t   = threadIdx.x;
  const float* pb = pw + ((size_t)b * S + pix) * INNER;
  float v0 = pb[t];
  float v1 = pb[t + 256];
  __shared__ float ssum[256], ssq[256];
  ssum[t] = v0 + v1;
  ssq[t]  = v0 * v0 + v1 * v1;
  __syncthreads();
  for (int off = 128; off > 0; off >>= 1) {
    if (t < off) { ssum[t] += ssum[t + off]; ssq[t] += ssq[t + off]; }
    __syncthreads();
  }
  float mu  = ssum[0] * (1.0f / 512.0f);
  float var = ssq[0] * (1.0f / 512.0f) - mu * mu;
  float inv = rsqrtf(var + 1e-5f);
  float n0 = (v0 - mu) * inv * gamma[t]       + beta[t];        // scales (c=t)
  float n1 = (v1 - mu) * inv * gamma[t + 256] + beta[t + 256];  // shifts
  size_t oi = ((size_t)b * C_IN + t) * S + pix;
  out[oi] = n1 + x[oi] * n0;
}

// ============================================================
extern "C" void kernel_launch(void* const* d_in, const int* in_sizes, int n_in,
                              void* d_out, int out_size, void* d_ws, size_t ws_size,
                              hipStream_t stream) {
  (void)in_sizes; (void)n_in; (void)out_size; (void)ws_size;
  const float* x     = (const float*)d_in[0];
  const float* w_in  = (const float*)d_in[1];
  const float* b_in  = (const float*)d_in[2];
  const float* lsc   = (const float*)d_in[3];
  const float* w_dw  = (const float*)d_in[4];
  const float* w_pw  = (const float*)d_in[5];
  const float* gamma = (const float*)d_in[6];
  const float* beta  = (const float*)d_in[7];
  float* out = (float*)d_out;

  const size_t nq = (size_t)BATCH * HEADS * S * DH;       // 2,359,296
  _Float16* qn  = (_Float16*)d_ws;
  _Float16* kn  = qn + nq;
  _Float16* vv  = kn + nq;          // [bh][d][s]
  float*    ao  = (float*)(vv + nq);
  _Float16* dwb = (_Float16*)(ao + nq);                   // [b][s][ch]
  float*    pwb = (float*)(dwb + (size_t)BATCH * INNER * S);  // [b][s][o]

  k_proj_in<<<BATCH * (OIN / 64) * (S / 64), 128, 0, stream>>>(x, w_in, b_in, qn, kn, vv);
  k_l2norm_k<<<(BATCH * HEADS * S) / 8, 256, 0, stream>>>(kn);
  k_attn<<<BATCH * HEADS * (S / 64), 128, 0, stream>>>(qn, kn, vv, lsc, ao);
  k_dw<<<(BATCH * INNER * S + 255) / 256, 256, 0, stream>>>(ao, w_dw, dwb);
  k_pw<<<BATCH * (INNER / 64) * (S / 64), 128, 0, stream>>>(dwb, w_pw, pwb);
  k_ln<<<BATCH * S, 256, 0, stream>>>(pwb, x, gamma, beta, out);
}
